// MultiHeadAttention_30863634989105
// MI455X (gfx1250) — matrix operand order repaired and checked by reading, not compiled, not run on hardware
//
#include <hip/hip_runtime.h>
#include <hip/hip_bf16.h>

typedef __attribute__((ext_vector_type(16))) _Float16 v16h;
typedef __attribute__((ext_vector_type(8)))  _Float16 v8h;
typedef __attribute__((ext_vector_type(4)))  _Float16 v4h;
typedef __attribute__((ext_vector_type(8)))  float    v8f;
typedef __attribute__((ext_vector_type(4)))  float    v4f;
typedef int v4i __attribute__((vector_size(16)));
typedef int v2i __attribute__((vector_size(8)));
typedef unsigned int u32x4 __attribute__((vector_size(16)));
typedef int i32x8 __attribute__((vector_size(32)));
typedef int i32x4 __attribute__((vector_size(16)));

#define B_   8
#define L_   1024
#define F_   1024
#define H_   16
#define D_   64
#define EPS_ 1e-5f

#if __has_builtin(__builtin_amdgcn_tensor_load_to_lds) && \
    __has_builtin(__builtin_amdgcn_s_wait_tensorcnt)
#define GEMM_TDM 1
#else
#define GEMM_TDM 0
#endif

// ---------------------------------------------------------------------------
// CDNA5 async global->LDS copies (ASYNCcnt-tracked), guarded for portability.
// Builtin signature (from hipcc diagnostics): (v4i AS1*, v4i AS3*, Ii, Ii).
// ---------------------------------------------------------------------------
static __device__ __forceinline__ void async_b128(const void* g, void* l) {
#if __has_builtin(__builtin_amdgcn_global_load_async_to_lds_b128)
  __builtin_amdgcn_global_load_async_to_lds_b128(
      (__attribute__((address_space(1))) v4i*)(void*)g,
      (__attribute__((address_space(3))) v4i*)(void*)l, 0, 0);
#else
  *(v8h*)l = *(const v8h*)g;
#endif
}

static __device__ __forceinline__ void async_b64(const void* g, void* l) {
#if __has_builtin(__builtin_amdgcn_global_load_async_to_lds_b64)
  __builtin_amdgcn_global_load_async_to_lds_b64(
      (__attribute__((address_space(1))) v2i*)(void*)g,
      (__attribute__((address_space(3))) v2i*)(void*)l, 0, 0);
#else
  *(v4h*)l = *(const v4h*)g;
#endif
}

static __device__ __forceinline__ void wait_async0() {
#if __has_builtin(__builtin_amdgcn_s_wait_asynccnt)
  __builtin_amdgcn_s_wait_asynccnt(0);
#endif
}

static __device__ __forceinline__ void wait_async2() {
#if __has_builtin(__builtin_amdgcn_s_wait_asynccnt)
  __builtin_amdgcn_s_wait_asynccnt(2);
#endif
}

#if GEMM_TDM
static __device__ __forceinline__ unsigned lds_off(void* p) {
  return (unsigned)(size_t)(__attribute__((address_space(3))) void*)p;
}

// Issue a 2-D TDM tile load: tile_d1 rows x tile_d0 f16 elements, row stride
// `stride` elements, into LDS at byte offset lds_addr. D# layout per CDNA5 ISA
// 8.3/8.4 (group0: count/lds/global/type; group1: data_size/dims/tile/stride).
// 6-arg builtin form: (u32x4 g0, i32x8 g1, i32x4 g2, i32x4 g3, i32x8, i32 cpol)
static __device__ __forceinline__ void tdm_load_2d(unsigned lds_addr,
                                                   unsigned long long gaddr,
                                                   int tile_d0, int tile_d1,
                                                   int stride) {
  u32x4 g0;
  g0[0] = 1u;                                            // count=1, user mode
  g0[1] = lds_addr;                                      // LDS byte address
  g0[2] = (unsigned)gaddr;                               // global_addr[31:0]
  g0[3] = ((unsigned)(gaddr >> 32) & 0x01FFFFFFu) |      // global_addr[56:32]
          0x80000000u;                                   // type=2 ("image")
  i32x8 g1;
  g1[0] = 1 << 16;                                       // data_size=1 (2B)
  g1[1] = (tile_d0 & 0xFFFF) << 16;                      // tensor_dim0 lo16
  g1[2] = ((tile_d0 >> 16) & 0xFFFF) |                   // tensor_dim0 hi16
          ((tile_d1 & 0xFFFF) << 16);                    // tensor_dim1 lo16
  g1[3] = ((tile_d1 >> 16) & 0xFFFF) |                   // tensor_dim1 hi16
          (tile_d0 << 16);                               // tile_dim0
  g1[4] = tile_d1;                                       // tile_dim1 (dim2=0)
  g1[5] = stride;                                        // dim0_stride lo32
  g1[6] = 0;                                             // stride hi / dim1_stride
  g1[7] = 0;
  const i32x4 z4 = {};
  const i32x8 z8 = {};
  __builtin_amdgcn_tensor_load_to_lds(g0, g1, z4, z4, z8, 0);
}

static __device__ __forceinline__ void wait_tensor0() {
  __builtin_amdgcn_s_wait_tensorcnt(0);
}
static __device__ __forceinline__ void wait_tensor2() {
  __builtin_amdgcn_s_wait_tensorcnt(2);
}
#endif

static __device__ __forceinline__ v16h mk16(v8h lo, v8h hi) {
  v16h r;
#pragma unroll
  for (int i = 0; i < 8; ++i) { r[i] = lo[i]; r[i + 8] = hi[i]; }
  return r;
}

static __device__ __forceinline__ v8f wmma16(v16h a, v16h b, v8f c) {
  // D = A(16x32 f16) * B(32x16 f16) + C(16x16 f32)
  return __builtin_amdgcn_wmma_f32_16x16x32_f16(false, a, false, b, (short)0, c,
                                                false, false);
}

// ---------------------------------------------------------------------------
// fp32 -> f16 conversion (makes GEMM staging a pure byte copy for the async /
// TDM engines). n must be a multiple of 1024.
// ---------------------------------------------------------------------------
__global__ __launch_bounds__(256) void to_half_kernel(
    const float* __restrict__ src, _Float16* __restrict__ dst) {
  const size_t i = ((size_t)blockIdx.x * 256 + threadIdx.x) * 4;
  const v4f x = *(const v4f*)(src + i);
  v4h h;
#pragma unroll
  for (int j = 0; j < 4; ++j) h[j] = (_Float16)x[j];
  *(v4h*)(dst + i) = h;
}

// ---------------------------------------------------------------------------
// GEMM: D[m,n] = sum_k A[m,k] * W[n,k] + bias[n]  (A,W in f16; 8192x1024x1024)
// Block tile 128(M) x 64(N), K-step 32, 8 waves x 4 WMMA. Tile staging via
// Tensor Data Mover (wave 0 issues 2 descriptors/stage, TENSORcnt-tracked),
// double-buffered; async-to-LDS fallback if TDM builtin is unavailable.
// ---------------------------------------------------------------------------
template <bool SCRAMBLE>
__global__ __launch_bounds__(256) void gemm_bias_kernel(
    const _Float16* __restrict__ A, const _Float16* __restrict__ W,
    const float* __restrict__ bias, void* __restrict__ Dptr) {
  __shared__ alignas(16) _Float16 As[2][128 * 32];  // 8192 B per buffer
  __shared__ alignas(16) _Float16 Bs[2][64 * 32];   // 4096 B per buffer

  const int t = threadIdx.x;
  const int mBase = blockIdx.y * 128;
  const int nBase = blockIdx.x * 64;
  const int wid = t >> 5, lane = t & 31, l16 = lane & 15, hilane = lane >> 4;
  const int waveM = (wid >> 1) * 32, waveN = (wid & 1) * 32;

  v8f acc[2][2];
#pragma unroll
  for (int i = 0; i < 2; ++i)
#pragma unroll
    for (int j = 0; j < 2; ++j) acc[i][j] = {};

  const int nK = F_ / 32;

#if GEMM_TDM
  const unsigned ldsA = lds_off(&As[0][0]);
  const unsigned ldsB = lds_off(&Bs[0][0]);
  const unsigned long long gA =
      (unsigned long long)(size_t)(A + (size_t)mBase * F_);
  const unsigned long long gB =
      (unsigned long long)(size_t)(W + (size_t)nBase * F_);
  if (wid == 0) {  // prologue: stage K-step 0 into buffer 0
    tdm_load_2d(ldsA, gA, 32, 128, F_);
    tdm_load_2d(ldsB, gB, 32, 64, F_);
  }
#else
  // staging assignments (uniform async count across waves)
  const int arow = t >> 1, acb = (t & 1) * 16;  // A: 2 thr/row, b128 each
  const int brow = t >> 2, bcb = (t & 3) * 8;   // B: 4 thr/row, b64 each
  const _Float16* Ag0 = A + (size_t)(mBase + arow) * F_ + acb;
  const _Float16* Wg0 = W + (size_t)(nBase + brow) * F_ + bcb;
  async_b128(Ag0, &As[0][arow * 32 + acb]);
  async_b64(Wg0, &Bs[0][brow * 32 + bcb]);
#endif

  for (int ks = 0; ks < nK; ++ks) {
#if GEMM_TDM
    if (wid == 0) {
      if (ks + 1 < nK) {
        const int nb = (ks + 1) & 1;
        const unsigned long long koff = (unsigned long long)(ks + 1) * 64u;
        tdm_load_2d(ldsA + nb * 8192u, gA + koff, 32, 128, F_);
        tdm_load_2d(ldsB + nb * 4096u, gB + koff, 32, 64, F_);
        wait_tensor2();  // retire stage ks (TDM in-order per wave)
      } else {
        wait_tensor0();
      }
    }
    __syncthreads();
#else
    if (ks + 1 < nK) {
      const int kn = (ks + 1) * 32;
      const int nb = (ks + 1) & 1;
      async_b128(Ag0 + kn, &As[nb][arow * 32 + acb]);
      async_b64(Wg0 + kn, &Bs[nb][brow * 32 + bcb]);
      wait_async2();  // retire stage ks (async loads complete in order)
    } else {
      wait_async0();
    }
    __syncthreads();
#endif

    const int cb = ks & 1;
    v16h a[2], b[2];
#pragma unroll
    for (int i = 0; i < 2; ++i) {
      const int r = waveM + 16 * i + l16, base = hilane * 8;
      a[i] = mk16(*(const v8h*)&As[cb][r * 32 + base],
                  *(const v8h*)&As[cb][r * 32 + base + 16]);
    }
#pragma unroll
    for (int j = 0; j < 2; ++j) {
      const int c = waveN + 16 * j + l16, base = hilane * 8;
      b[j] = mk16(*(const v8h*)&Bs[cb][c * 32 + base],
                  *(const v8h*)&Bs[cb][c * 32 + base + 16]);
    }
#pragma unroll
    for (int i = 0; i < 2; ++i)
#pragma unroll
      for (int j = 0; j < 2; ++j) acc[i][j] = wmma16(a[i], b[j], acc[i][j]);
    __syncthreads();
  }

#pragma unroll
  for (int i = 0; i < 2; ++i)
#pragma unroll
    for (int j = 0; j < 2; ++j)
#pragma unroll
      for (int r = 0; r < 8; ++r) {
        const int gm = mBase + waveM + 16 * i + r + hilane * 8;
        const int gn = nBase + waveN + 16 * j + l16;
        const float v = acc[i][j][r] + bias[gn];
        if (SCRAMBLE) {
          const int bb = gm >> 10, l = gm & (L_ - 1);
          const int h = gn >> 6, d = gn & (D_ - 1);
          ((_Float16*)Dptr)[(((size_t)(bb * H_ + h)) * L_ + l) * D_ + d] =
              (_Float16)v;
        } else {
          ((float*)Dptr)[(size_t)gm * F_ + gn] = v;
        }
      }
}

// ---------------------------------------------------------------------------
// Attention: per block = one (head n, 16 query rows).
//   S = Qtile(16x64) @ Kh^T -> softmax (1024-wide rows) -> raw_att store
//   att = P(16x1024) @ Vh(1024x64), V transposed through LDS for B-operand.
// ---------------------------------------------------------------------------
__global__ __launch_bounds__(256) void attention_kernel(
    const _Float16* __restrict__ Qh, const _Float16* __restrict__ Kh,
    const _Float16* __restrict__ Vh, float* __restrict__ ratt,
    _Float16* __restrict__ attc) {
  extern __shared__ char smem[];
  float* S = (float*)smem;                // 16*1024 f32  (65536 B)
  float* red = S + 16 * 1024;             // 16*16        (1024 B)
  float* rs = red + 256;                  // 16           (64 B)
  float* pac = rs + 16;                   // 4*16*16      (4096 B)
  _Float16* P = (_Float16*)(pac + 1024);  // 16*1024 f16  (32768 B)
  _Float16* Qs = P + 16 * 1024;           // 16*64        (2048 B)
  _Float16* Vt = Qs + 16 * 64;            // 64*64        (8192 B)

  const int t = threadIdx.x;
  const int q0 = blockIdx.x * 16;
  const int n = blockIdx.y;
  const int wid = t >> 5, lane = t & 31, l16 = lane & 15, hilane = lane >> 4;

  // async-stage 16x64 Q tile to LDS (64 b128 chunks, threads 0..63)
  if (t < 64) {
    const int r = t >> 2, cb = (t & 3) * 16;
    async_b128(Qh + ((size_t)n * L_ + q0 + r) * D_ + cb, &Qs[r * 64 + cb]);
  }
  wait_async0();
  __syncthreads();

  v16h aq[2];
#pragma unroll
  for (int s = 0; s < 2; ++s) {
    const int base = s * 32 + hilane * 8;
    aq[s] = mk16(*(const v8h*)&Qs[l16 * 64 + base],
                 *(const v8h*)&Qs[l16 * 64 + base + 16]);
  }

  // Phase 1: S = Q K^T * 1/sqrt(D); each wave covers 8 of 64 key tiles.
  for (int kt = wid; kt < 64; kt += 8) {
    const int n0 = kt * 16;
    const _Float16* kr = Kh + ((size_t)n * L_ + n0 + l16) * D_;
    v8f acc = {};
#pragma unroll
    for (int s = 0; s < 2; ++s) {
      const int base = s * 32 + hilane * 8;
      const v16h bk =
          mk16(*(const v8h*)(kr + base), *(const v8h*)(kr + base + 16));
      acc = wmma16(aq[s], bk, acc);
    }
#pragma unroll
    for (int r = 0; r < 8; ++r)
      S[(r + hilane * 8) * 1024 + n0 + l16] = acc[r] * 0.125f;
  }
  __syncthreads();

  // Phase 2: row softmax (row = t>>4, 64-col segment = t&15).
  {
    const int row = t >> 4, seg = t & 15;
    float* sr = S + row * 1024 + seg * 64;
    float m = -3.0e38f;
#pragma unroll 8
    for (int c = 0; c < 64; ++c) m = fmaxf(m, sr[c]);
    red[row * 16 + seg] = m;
    __syncthreads();
    if (seg == 0) {
      float mm = red[row * 16];
#pragma unroll
      for (int x = 1; x < 16; ++x) mm = fmaxf(mm, red[row * 16 + x]);
      rs[row] = mm;
    }
    __syncthreads();
    const float mm = rs[row];
    float sum = 0.f;
#pragma unroll 8
    for (int c = 0; c < 64; ++c) {
      const float e = __expf(sr[c] - mm);
      sr[c] = e;
      sum += e;
    }
    red[row * 16 + seg] = sum;
    __syncthreads();
    if (seg == 0) {
      float ss = 0.f;
#pragma unroll
      for (int x = 0; x < 16; ++x) ss += red[row * 16 + x];
      rs[row] = 1.0f / ss;
    }
    __syncthreads();
    const float inv = rs[row];
    float* gp = ratt + ((size_t)n * L_ + q0 + row) * L_ + seg * 64;
    _Float16* pp = P + row * 1024 + seg * 64;
#pragma unroll
    for (int c4 = 0; c4 < 16; ++c4) {
      v4f pv = *(const v4f*)(sr + c4 * 4);
      v4h ph;
#pragma unroll
      for (int x = 0; x < 4; ++x) {
        pv[x] *= inv;
        ph[x] = (_Float16)pv[x];
      }
      *(v4f*)(gp + c4 * 4) = pv;  // raw_att output (fp32)
      *(v4h*)(pp + c4 * 4) = ph;
    }
  }
  __syncthreads();

  // Phase 3: att = P @ V. 16 stages of 64 K-rows; waves: 4 n-tiles x 2 k-halves.
  const int ntile = wid & 3, ksub = wid >> 2;
  v8f accv = {};
  for (int stage = 0; stage < 16; ++stage) {
    const int kb = stage * 64;
    {  // stage V transposed: Vt[d][k]
      const int krow = t >> 2, cb = (t & 3) * 16;
      const _Float16* vsrc = Vh + ((size_t)n * L_ + kb + krow) * D_ + cb;
      const v8h lo = *(const v8h*)vsrc, hi8 = *(const v8h*)(vsrc + 8);
#pragma unroll
      for (int x = 0; x < 8; ++x) {
        Vt[(cb + x) * 64 + krow] = lo[x];
        Vt[(cb + 8 + x) * 64 + krow] = hi8[x];
      }
    }
    __syncthreads();
    {
      const int k32 = ksub * 32;
      const int base = kb + k32 + hilane * 8;
      const v16h ap = mk16(*(const v8h*)&P[l16 * 1024 + base],
                           *(const v8h*)&P[l16 * 1024 + base + 16]);
      const int d = ntile * 16 + l16, vb = k32 + hilane * 8;
      const v16h bv = mk16(*(const v8h*)&Vt[d * 64 + vb],
                           *(const v8h*)&Vt[d * 64 + vb + 16]);
      accv = wmma16(ap, bv, accv);
    }
    __syncthreads();
  }
  if (ksub == 1) {
#pragma unroll
    for (int r = 0; r < 8; ++r)
      pac[(ntile * 16 + r + hilane * 8) * 16 + l16] = accv[r];
  }
  __syncthreads();
  if (ksub == 0) {
    // faithful recombine: att.reshape(H,B,L,D).transpose(1,2,0,3)
    const int b2 = n & (B_ - 1), h2 = n >> 3;
#pragma unroll
    for (int r = 0; r < 8; ++r) {
      const int M = r + hilane * 8;
      const float v = accv[r] + pac[(ntile * 16 + M) * 16 + l16];
      const int d = ntile * 16 + l16;
      attc[((size_t)b2 * L_ + q0 + M) * F_ + h2 * D_ + d] = (_Float16)v;
    }
  }
}

// ---------------------------------------------------------------------------
// LayerNorm: block per row; x = proj + residual; mean/var over F=1024.
// ---------------------------------------------------------------------------
__global__ __launch_bounds__(256) void layernorm_kernel(
    const float* __restrict__ tmp, const float* __restrict__ resid,
    const float* __restrict__ gamma, const float* __restrict__ beta,
    float* __restrict__ out) {
  __shared__ float rbuf[256];
  const int m = blockIdx.x, t = threadIdx.x;
  const float* tr = tmp + (size_t)m * F_;
  const float* qr = resid + (size_t)m * F_;
  float x[4];
  float s = 0.f;
#pragma unroll
  for (int j = 0; j < 4; ++j) {
    const int i = t + j * 256;
    x[j] = tr[i] + qr[i];
    s += x[j];
  }
  rbuf[t] = s;
  __syncthreads();
  for (int off = 128; off > 0; off >>= 1) {
    if (t < off) rbuf[t] += rbuf[t + off];
    __syncthreads();
  }
  const float mu = rbuf[0] * (1.0f / F_);
  __syncthreads();
  float v = 0.f;
#pragma unroll
  for (int j = 0; j < 4; ++j) {
    const float d = x[j] - mu;
    v += d * d;
  }
  rbuf[t] = v;
  __syncthreads();
  for (int off = 128; off > 0; off >>= 1) {
    if (t < off) rbuf[t] += rbuf[t + off];
    __syncthreads();
  }
  const float inv = rsqrtf(rbuf[0] * (1.0f / F_) + EPS_);
#pragma unroll
  for (int j = 0; j < 4; ++j) {
    const int i = t + j * 256;
    out[(size_t)m * F_ + i] = (x[j] - mu) * inv * gamma[i] + beta[i];
  }
}

extern "C" void kernel_launch(void* const* d_in, const int* in_sizes, int n_in,
                              void* d_out, int out_size, void* d_ws,
                              size_t ws_size, hipStream_t stream) {
  (void)in_sizes; (void)n_in; (void)out_size; (void)ws_size;
  const float* q = (const float*)d_in[0];
  const float* k = (const float*)d_in[1];
  const float* v = (const float*)d_in[2];
  const float* Wq = (const float*)d_in[3];
  const float* bq = (const float*)d_in[4];
  const float* Wk = (const float*)d_in[5];
  const float* bk = (const float*)d_in[6];
  const float* Wv = (const float*)d_in[7];
  const float* bv = (const float*)d_in[8];
  const float* Wf = (const float*)d_in[9];
  const float* bf = (const float*)d_in[10];
  const float* gamma = (const float*)d_in[11];
  const float* beta = (const float*)d_in[12];

  float* out = (float*)d_out;
  float* ratt = out + (size_t)B_ * L_ * F_;  // second tuple output

  const size_t NE = (size_t)B_ * L_ * F_;  // 8M elements
  _Float16* X = (_Float16*)d_ws;           // reused f16 activation copy
  _Float16* Wb = X + NE;                   // reused f16 weight copy (1M)
  _Float16* Qh = Wb + (size_t)F_ * F_;
  _Float16* Kh = Qh + NE;
  _Float16* Vh = Kh + NE;
  _Float16* Attc = Vh + NE;
  float* Tmp = (float*)(Attc + NE);

  const dim3 gg(F_ / 64, (B_ * L_) / 128);
  const int gAct = (int)(NE / 1024);               // 8192 blocks
  const int gW = (int)(((size_t)F_ * F_) / 1024);  // 1024 blocks

  // Q projection
  to_half_kernel<<<gAct, 256, 0, stream>>>(q, X);
  to_half_kernel<<<gW, 256, 0, stream>>>(Wq, Wb);
  gemm_bias_kernel<true><<<gg, 256, 0, stream>>>(X, Wb, bq, Qh);
  // K projection
  to_half_kernel<<<gAct, 256, 0, stream>>>(k, X);
  to_half_kernel<<<gW, 256, 0, stream>>>(Wk, Wb);
  gemm_bias_kernel<true><<<gg, 256, 0, stream>>>(X, Wb, bk, Kh);
  // V projection
  to_half_kernel<<<gAct, 256, 0, stream>>>(v, X);
  to_half_kernel<<<gW, 256, 0, stream>>>(Wv, Wb);
  gemm_bias_kernel<true><<<gg, 256, 0, stream>>>(X, Wb, bv, Vh);

  const size_t smem = 16 * 1024 * 4 + 256 * 4 + 16 * 4 + 1024 * 4 +
                      16 * 1024 * 2 + 16 * 64 * 2 + 64 * 64 * 2;  // 113728 B
  (void)hipFuncSetAttribute((const void*)attention_kernel,
                            hipFuncAttributeMaxDynamicSharedMemorySize,
                            (int)smem);
  attention_kernel<<<dim3(L_ / 16, B_ * H_), 256, smem, stream>>>(Qh, Kh, Vh,
                                                                  ratt, Attc);

  // output projection + layernorm
  to_half_kernel<<<gW, 256, 0, stream>>>(Wf, Wb);
  gemm_bias_kernel<false><<<gg, 256, 0, stream>>>(Attc, Wb, bf, Tmp);
  layernorm_kernel<<<B_ * L_, 256, 0, stream>>>(Tmp, q, gamma, beta, out);
}
